// CAM_Module_1726576855038
// MI455X (gfx1250) — matrix operand order repaired and checked by reading, not compiled, not run on hardware
//
#include <hip/hip_runtime.h>

// Problem dims (fixed by the reference): x is (B=16, C=512, H=64, W=64), N=H*W=4096
#define B_DIM 16
#define C_DIM 512
#define N_DIM 4096

typedef __attribute__((ext_vector_type(16))) _Float16     v16h;
typedef __attribute__((ext_vector_type(8)))  float        v8f;
typedef __attribute__((ext_vector_type(4)))  unsigned int u32x4;

union F16Frag { v16h v; u32x4 q[2]; };
union F16Pack { _Float16 h[2]; unsigned u; };

// ---------------------------------------------------------------------------
// Fragment loaders matching CDNA5 ISA 16-bit WMMA VGPR layouts (wave32).
// A (16x32, MxK): lane l -> row m=l%16, half h=l/16; K chunks {h*8..h*8+7} and
//                 {16+h*8..16+h*8+7} (two 16B contiguous loads).
// B (32x16, KxN): lane l -> col n=l%16, half h=l/16; K chunks {h*8..h*8+7} and
//                 {16+h*8..16+h*8+7} (two 16B contiguous loads). Source rows supply K contiguously,
//                 so B is read from rows of the (pre)transposed operand.
// ---------------------------------------------------------------------------
__device__ __forceinline__ v16h load_a_frag(const _Float16* rowbase, int ldk, int k, int lane) {
    const int m = lane & 15;
    const int h = lane >> 4;
    const _Float16* p = rowbase + (size_t)m * ldk + k + h * 8;
    F16Frag f;
    f.q[0] = *(const u32x4*)(p);
    f.q[1] = *(const u32x4*)(p + 16);
    return f.v;
}

__device__ __forceinline__ v16h load_b_frag(const _Float16* rowbase, int ldk, int k, int lane) {
    const int n = lane & 15;
    const int h = lane >> 4;
    const _Float16* p = rowbase + (size_t)n * ldk + k + h * 8;
    F16Frag f;
    f.q[0] = *(const u32x4*)(p);
    f.q[1] = *(const u32x4*)(p + 16);
    return f.v;
}

// ---------------------------------------------------------------------------
// Kernel 1: convert x (f32, B*C*N) -> V (f16, row-major C x N per batch)
//           and VT (f16, row-major N x C per batch) via LDS tile transpose.
// Grid: (N/64, C/32, B), block: (32, 8). All global stores are packed 4B.
// ---------------------------------------------------------------------------
__global__ __launch_bounds__(256) void cvt_transpose_kernel(const float* __restrict__ x,
                                                            _Float16* __restrict__ V,
                                                            _Float16* __restrict__ VT) {
    __shared__ _Float16 tile[64][33];   // [n_local][c_local], padded
    const int b  = blockIdx.z;
    const int c0 = blockIdx.y * 32;
    const int n0 = blockIdx.x * 64;
    const int tx = threadIdx.x;         // 0..31
    const int ty = threadIdx.y;         // 0..7
    const float* xb = x  + (size_t)b * C_DIM * N_DIM;
    _Float16*    Vb = V  + (size_t)b * C_DIM * N_DIM;
    _Float16*    Tb = VT + (size_t)b * N_DIM * C_DIM;

#pragma unroll
    for (int r = 0; r < 4; ++r) {
        const int c = ty + r * 8;
        const float2 f2 = *(const float2*)(xb + (size_t)(c0 + c) * N_DIM + n0 + 2 * tx);
        F16Pack pk;
        pk.h[0] = (_Float16)f2.x;
        pk.h[1] = (_Float16)f2.y;
        *(unsigned*)(Vb + (size_t)(c0 + c) * N_DIM + n0 + 2 * tx) = pk.u;
        tile[2 * tx][c]     = pk.h[0];
        tile[2 * tx + 1][c] = pk.h[1];
    }
    __syncthreads();

    const int t = ty * 32 + tx;         // 0..255
#pragma unroll
    for (int r = 0; r < 4; ++r) {
        const int idx = r * 256 + t;    // 0..1023
        const int nl  = idx >> 4;       // 0..63
        const int cp  = idx & 15;       // column pair -> c = 2*cp
        F16Pack pk;
        pk.h[0] = tile[nl][2 * cp];
        pk.h[1] = tile[nl][2 * cp + 1];
        *(unsigned*)(Tb + (size_t)(n0 + nl) * C_DIM + c0 + 2 * cp) = pk.u;
    }
}

// ---------------------------------------------------------------------------
// Kernel 2: energy[b] = V[b] * V[b]^T  (512x512, K=4096), f16 in / f32 out.
// Grid: (C/128, C/128, B), block 256 threads = 8 waves in 2(M) x 4(N).
// Each wave computes a 64x32 tile = 4x2 WMMA accumulators.
// ---------------------------------------------------------------------------
__global__ __launch_bounds__(256) void gram_kernel(const _Float16* __restrict__ V,
                                                   float* __restrict__ energy) {
    const int b    = blockIdx.z;
    const int lane = threadIdx.x & 31;
    const int wave = threadIdx.x >> 5;
    const int wm   = wave >> 2;   // 0..1
    const int wn   = wave & 3;    // 0..3
    const int m0   = blockIdx.y * 128 + wm * 64;
    const int n0   = blockIdx.x * 128 + wn * 32;
    const _Float16* Vb = V + (size_t)b * C_DIM * N_DIM;

    v8f acc[4][2] = {};

    for (int k = 0; k < N_DIM; k += 32) {
        v16h afr[4], bfr[2];
#pragma unroll
        for (int mt = 0; mt < 4; ++mt)
            afr[mt] = load_a_frag(Vb + (size_t)(m0 + 16 * mt) * N_DIM, N_DIM, k, lane);
#pragma unroll
        for (int nt = 0; nt < 2; ++nt)
            bfr[nt] = load_b_frag(Vb + (size_t)(n0 + 16 * nt) * N_DIM, N_DIM, k, lane);
#pragma unroll
        for (int mt = 0; mt < 4; ++mt)
#pragma unroll
            for (int nt = 0; nt < 2; ++nt)
                acc[mt][nt] = __builtin_amdgcn_wmma_f32_16x16x32_f16(
                    false, afr[mt], false, bfr[nt], (short)0, acc[mt][nt], false, false);
    }

    // C/D layout: VGPR g <-> M = g + 8*(lane/16), N = lane%16
    float* Eb = energy + (size_t)b * C_DIM * C_DIM;
    const int col = lane & 15;
    const int h   = lane >> 4;
#pragma unroll
    for (int mt = 0; mt < 4; ++mt)
#pragma unroll
        for (int nt = 0; nt < 2; ++nt)
#pragma unroll
            for (int g = 0; g < 8; ++g)
                Eb[(size_t)(m0 + 16 * mt + g + 8 * h) * C_DIM + (n0 + 16 * nt + col)] =
                    acc[mt][nt][g];
}

// ---------------------------------------------------------------------------
// Kernel 3: att = softmax(rowmax - energy) == exp(rowmin - e)/sum(exp(rowmin - e))
// One block (256 thr) per row of 512; writes att in f16 for the second GEMM.
// ---------------------------------------------------------------------------
__global__ __launch_bounds__(256) void softmax_kernel(const float* __restrict__ energy,
                                                      _Float16* __restrict__ att) {
    __shared__ float red[256];
    const int t = threadIdx.x;
    const float* row = energy + (size_t)blockIdx.x * C_DIM;
    const float v0 = row[t];
    const float v1 = row[t + 256];

    red[t] = fminf(v0, v1);
    __syncthreads();
#pragma unroll
    for (int s = 128; s > 0; s >>= 1) {
        if (t < s) red[t] = fminf(red[t], red[t + s]);
        __syncthreads();
    }
    const float rmin = red[0];
    __syncthreads();

    const float e0 = __expf(rmin - v0);
    const float e1 = __expf(rmin - v1);
    red[t] = e0 + e1;
    __syncthreads();
#pragma unroll
    for (int s = 128; s > 0; s >>= 1) {
        if (t < s) red[t] += red[t + s];
        __syncthreads();
    }
    const float inv = 1.0f / red[0];

    _Float16* orow = att + (size_t)blockIdx.x * C_DIM;
    orow[t]       = (_Float16)(e0 * inv);
    orow[t + 256] = (_Float16)(e1 * inv);
}

// ---------------------------------------------------------------------------
// Kernel 4: out = gamma * (att @ V) + x   (M=512, N=4096, K=512 per batch)
// A = att rows (f16), B fragments from VT rows (f16, contiguous in K).
// Grid: (N/128, C/128, B), block 256 threads = 8 waves (2x4), wave tile 64x32.
// ---------------------------------------------------------------------------
__global__ __launch_bounds__(256) void out_kernel(const _Float16* __restrict__ att,
                                                  const _Float16* __restrict__ VT,
                                                  const float* __restrict__ x,
                                                  const float* __restrict__ gamma,
                                                  float* __restrict__ out) {
    const int b    = blockIdx.z;
    const int lane = threadIdx.x & 31;
    const int wave = threadIdx.x >> 5;
    const int wm   = wave >> 2;
    const int wn   = wave & 3;
    const int m0   = blockIdx.y * 128 + wm * 64;   // channel (i)
    const int n0   = blockIdx.x * 128 + wn * 32;   // spatial (n)
    const _Float16* Ab = att + (size_t)b * C_DIM * C_DIM;
    const _Float16* Tb = VT  + (size_t)b * N_DIM * C_DIM;

    v8f acc[4][2] = {};

    for (int k = 0; k < C_DIM; k += 32) {
        v16h afr[4], bfr[2];
#pragma unroll
        for (int mt = 0; mt < 4; ++mt)
            afr[mt] = load_a_frag(Ab + (size_t)(m0 + 16 * mt) * C_DIM, C_DIM, k, lane);
#pragma unroll
        for (int nt = 0; nt < 2; ++nt)
            bfr[nt] = load_b_frag(Tb + (size_t)(n0 + 16 * nt) * C_DIM, C_DIM, k, lane);
#pragma unroll
        for (int mt = 0; mt < 4; ++mt)
#pragma unroll
            for (int nt = 0; nt < 2; ++nt)
                acc[mt][nt] = __builtin_amdgcn_wmma_f32_16x16x32_f16(
                    false, afr[mt], false, bfr[nt], (short)0, acc[mt][nt], false, false);
    }

    const float g = gamma[0];
    const float* xb = x   + (size_t)b * C_DIM * N_DIM;
    float*       ob = out + (size_t)b * C_DIM * N_DIM;
    const int col = lane & 15;
    const int h   = lane >> 4;
#pragma unroll
    for (int mt = 0; mt < 4; ++mt)
#pragma unroll
        for (int nt = 0; nt < 2; ++nt)
#pragma unroll
            for (int gi = 0; gi < 8; ++gi) {
                const size_t idx =
                    (size_t)(m0 + 16 * mt + gi + 8 * h) * N_DIM + (n0 + 16 * nt + col);
                ob[idx] = g * acc[mt][nt][gi] + xb[idx];
            }
}

// ---------------------------------------------------------------------------
extern "C" void kernel_launch(void* const* d_in, const int* in_sizes, int n_in,
                              void* d_out, int out_size, void* d_ws, size_t ws_size,
                              hipStream_t stream) {
    const float* x     = (const float*)d_in[0];
    const float* gamma = (const float*)d_in[1];
    float*       out   = (float*)d_out;

    char* ws = (char*)d_ws;
    const size_t vbytes = (size_t)B_DIM * C_DIM * N_DIM * sizeof(_Float16); // 67,108,864
    const size_t ebytes = (size_t)B_DIM * C_DIM * C_DIM * sizeof(float);    // 16,777,216
    _Float16* V      = (_Float16*)(ws);
    _Float16* VT     = (_Float16*)(ws + vbytes);
    float*    energy = (float*)(ws + 2 * vbytes);
    _Float16* att    = (_Float16*)(ws + 2 * vbytes + ebytes);

    cvt_transpose_kernel<<<dim3(N_DIM / 64, C_DIM / 32, B_DIM), dim3(32, 8), 0, stream>>>(x, V, VT);
    gram_kernel<<<dim3(C_DIM / 128, C_DIM / 128, B_DIM), 256, 0, stream>>>(V, energy);
    softmax_kernel<<<dim3(B_DIM * C_DIM), 256, 0, stream>>>(energy, att);
    out_kernel<<<dim3(N_DIM / 128, C_DIM / 128, B_DIM), 256, 0, stream>>>(att, VT, x, gamma, out);
}
